// TokenDiscrepancyLoss_62972810494575
// MI455X (gfx1250) — matrix operand order repaired and checked by reading, not compiled, not run on hardware
//
#include <hip/hip_runtime.h>
#include <hip/hip_bf16.h>

// ---------------------------------------------------------------------------
// TokenDiscrepancyLoss fused kernel for gfx1250 (MI455X)
//   B=4, S=2048 -> 8192 tokens; H=1024; C=8192
//   loss = 0.1 * sum_masked( sum_c softmax(hW+b)_c * (||t||^2 - 2 t.cb_c + ||cb_c||^2)/H )
// bf16 WMMA 16x16x32 for both GEMMs, 2M x 2N register blocking, streaming
// (online) softmax so no [B,S,C] tensor is materialized.
// ---------------------------------------------------------------------------

typedef __attribute__((ext_vector_type(16))) __bf16 v16bf;
typedef __attribute__((ext_vector_type(8)))  __bf16 v8bf;
typedef __attribute__((ext_vector_type(4)))  __bf16 v4bf;
typedef __attribute__((ext_vector_type(8)))  float  v8f;

#define HIDDEN   1024
#define CODEBOOK 8192
#define NTOK     8192   /* 4 * 2048 */

__global__ void k_zero(float* out) { out[0] = 0.f; }

// [rows, HIDDEN] f32 -> bf16 row-major + per-row sum of squares.
__global__ void __launch_bounds__(256)
k_row_convert(const float* __restrict__ src, __bf16* __restrict__ dst,
              float* __restrict__ sq) {
    const int row = blockIdx.x;
    const float* s = src + (size_t)row * HIDDEN;
    __bf16*      d = dst + (size_t)row * HIDDEN;
    float acc = 0.f;
    for (int i = threadIdx.x; i < HIDDEN; i += 256) {
        float v = s[i];
        d[i] = (__bf16)v;
        acc += v * v;
    }
    acc += __shfl_xor(acc, 16, 32);
    acc += __shfl_xor(acc,  8, 32);
    acc += __shfl_xor(acc,  4, 32);
    acc += __shfl_xor(acc,  2, 32);
    acc += __shfl_xor(acc,  1, 32);
    __shared__ float part[8];
    const int lane = threadIdx.x & 31, w = threadIdx.x >> 5;
    if (lane == 0) part[w] = acc;
    __syncthreads();
    if (threadIdx.x == 0) {
        float t = 0.f;
        for (int i = 0; i < 8; ++i) t += part[i];
        sq[row] = t;
    }
}

// Per-row sum of squares only (for target: bf16 copy not needed, A is staged in LDS).
__global__ void __launch_bounds__(256)
k_row_sq(const float* __restrict__ src, float* __restrict__ sq) {
    const int row = blockIdx.x;
    const float* s = src + (size_t)row * HIDDEN;
    float acc = 0.f;
    for (int i = threadIdx.x; i < HIDDEN; i += 256) {
        float v = s[i];
        acc += v * v;
    }
    acc += __shfl_xor(acc, 16, 32);
    acc += __shfl_xor(acc,  8, 32);
    acc += __shfl_xor(acc,  4, 32);
    acc += __shfl_xor(acc,  2, 32);
    acc += __shfl_xor(acc,  1, 32);
    __shared__ float part[8];
    const int lane = threadIdx.x & 31, w = threadIdx.x >> 5;
    if (lane == 0) part[w] = acc;
    __syncthreads();
    if (threadIdx.x == 0) {
        float t = 0.f;
        for (int i = 0; i < 8; ++i) t += part[i];
        sq[row] = t;
    }
}

// W [HIDDEN, CODEBOOK] f32 -> Wt [CODEBOOK, HIDDEN] bf16 (32x32 LDS tile transpose).
__global__ void __launch_bounds__(256)
k_transpose_w(const float* __restrict__ W, __bf16* __restrict__ Wt) {
    __shared__ float tile[32][33];
    const int c0 = blockIdx.x * 32, h0 = blockIdx.y * 32;
    const int tx = threadIdx.x & 31, ty = threadIdx.x >> 5;   // 32 x 8
#pragma unroll
    for (int r = 0; r < 4; ++r) {
        int hh = ty + r * 8;
        tile[hh][tx] = W[(size_t)(h0 + hh) * CODEBOOK + (c0 + tx)];
    }
    __syncthreads();
#pragma unroll
    for (int r = 0; r < 4; ++r) {
        int cc = ty + r * 8;
        Wt[(size_t)(c0 + cc) * HIDDEN + (h0 + tx)] = (__bf16)tile[tx][cc];
    }
}

// Main fused kernel: one block = 32 tokens (2 M-tiles); 8 waves each stream
// 64 codebook tiles, 2 N-tiles at a time (2M x 2N register blocking).
__global__ void __launch_bounds__(256)
k_main(const float* __restrict__ hidden, const int* __restrict__ ttype,
       const float* __restrict__ target,
       const __bf16* __restrict__ cbb, const __bf16* __restrict__ Wt,
       const float* __restrict__ bias,
       const float* __restrict__ csq,  const float* __restrict__ tsq,
       float* __restrict__ out) {
    __shared__ __align__(32) __bf16 Ah[32 * HIDDEN];   // 64 KB hidden A (bf16)
    __shared__ __align__(32) __bf16 At[32 * HIDDEN];   // 64 KB target A (bf16)
    __shared__ float tsq_s[32];
    __shared__ float red[8][32][3];
    __shared__ float fin[32];

    const int m0 = blockIdx.x * 32;

    // Stage + convert both A tiles: 32 tokens x 1024, float4 reads, ds_store_b64.
    {
        const float4* hs = (const float4*)(hidden + (size_t)m0 * HIDDEN);
        const float4* ts = (const float4*)(target + (size_t)m0 * HIDDEN);
        for (int i = threadIdx.x; i < 32 * HIDDEN / 4; i += 256) {
            float4 hv = hs[i];
            float4 tv = ts[i];
            *(v4bf*)(Ah + 4 * i) = (v4bf){(__bf16)hv.x, (__bf16)hv.y, (__bf16)hv.z, (__bf16)hv.w};
            *(v4bf*)(At + 4 * i) = (v4bf){(__bf16)tv.x, (__bf16)tv.y, (__bf16)tv.z, (__bf16)tv.w};
        }
        if (threadIdx.x < 32) tsq_s[threadIdx.x] = tsq[m0 + threadIdx.x];
    }
    __syncthreads();

    const int lane = threadIdx.x & 31;
    const int w    = threadIdx.x >> 5;
    const int l15  = lane & 15;
    const int half = lane >> 4;

    // A fragments (contiguous k order used here: lane<16 -> K 0..15; lane>=16 -> K 16..31).
    const __bf16* ah0 = Ah + (     l15) * HIDDEN + half * 16;   // rows  0..15
    const __bf16* ah1 = Ah + (16 + l15) * HIDDEN + half * 16;   // rows 16..31
    const __bf16* at0 = At + (     l15) * HIDDEN + half * 16;
    const __bf16* at1 = At + (16 + l15) * HIDDEN + half * 16;

    float tsqr[2][8];
#pragma unroll
    for (int mi = 0; mi < 2; ++mi)
#pragma unroll
        for (int j = 0; j < 8; ++j) tsqr[mi][j] = tsq_s[mi * 16 + half * 8 + j];

    float m_run[2][8], s_run[2][8], n_run[2][8];
#pragma unroll
    for (int mi = 0; mi < 2; ++mi)
#pragma unroll
        for (int j = 0; j < 8; ++j) { m_run[mi][j] = -3.0e38f; s_run[mi][j] = 0.f; n_run[mi][j] = 0.f; }

    for (int t = 0; t < 32; ++t) {
        const int c0 = (w * 64 + 2 * t) * 16;          // 2 N-tiles: codes c0..c0+31
        // B as used here: lane = N column; same contiguous k order as A (lanes 0..15 K 0..15, 16..31 K 16..31).
        const __bf16* bw = Wt  + (size_t)(c0 + l15) * HIDDEN + half * 16;
        const __bf16* bc = cbb + (size_t)(c0 + l15) * HIDDEN + half * 16;

        v8f aL[2][2] = {{{}, {}}, {{}, {}}};   // [mi][ni] logits accumulators
        v8f aX[2][2] = {{{}, {}}, {{}, {}}};   // [mi][ni] cross accumulators
#pragma unroll 2
        for (int kc = 0; kc < HIDDEN; kc += 32) {
            v8bf h00 = *(const v8bf*)(ah0 + kc);
            v8bf h01 = *(const v8bf*)(ah0 + kc + 8);
            v8bf h10 = *(const v8bf*)(ah1 + kc);
            v8bf h11 = *(const v8bf*)(ah1 + kc + 8);
            v8bf t00 = *(const v8bf*)(at0 + kc);
            v8bf t01 = *(const v8bf*)(at0 + kc + 8);
            v8bf t10 = *(const v8bf*)(at1 + kc);
            v8bf t11 = *(const v8bf*)(at1 + kc + 8);
            v16bf ahf0, ahf1, atf0, atf1;
#pragma unroll
            for (int i = 0; i < 8; ++i) {
                ahf0[i] = h00[i]; ahf0[8 + i] = h01[i];
                ahf1[i] = h10[i]; ahf1[8 + i] = h11[i];
                atf0[i] = t00[i]; atf0[8 + i] = t01[i];
                atf1[i] = t10[i]; atf1[8 + i] = t11[i];
            }
            v16bf bw0 = *(const v16bf*)(bw + kc);
            v16bf bw1 = *(const v16bf*)(bw + (size_t)16 * HIDDEN + kc);
            v16bf bc0 = *(const v16bf*)(bc + kc);
            v16bf bc1 = *(const v16bf*)(bc + (size_t)16 * HIDDEN + kc);

            aL[0][0] = __builtin_amdgcn_wmma_f32_16x16x32_bf16(false, ahf0, false, bw0, (short)0, aL[0][0], false, false);
            aL[1][0] = __builtin_amdgcn_wmma_f32_16x16x32_bf16(false, ahf1, false, bw0, (short)0, aL[1][0], false, false);
            aL[0][1] = __builtin_amdgcn_wmma_f32_16x16x32_bf16(false, ahf0, false, bw1, (short)0, aL[0][1], false, false);
            aL[1][1] = __builtin_amdgcn_wmma_f32_16x16x32_bf16(false, ahf1, false, bw1, (short)0, aL[1][1], false, false);
            aX[0][0] = __builtin_amdgcn_wmma_f32_16x16x32_bf16(false, atf0, false, bc0, (short)0, aX[0][0], false, false);
            aX[1][0] = __builtin_amdgcn_wmma_f32_16x16x32_bf16(false, atf1, false, bc0, (short)0, aX[1][0], false, false);
            aX[0][1] = __builtin_amdgcn_wmma_f32_16x16x32_bf16(false, atf0, false, bc1, (short)0, aX[0][1], false, false);
            aX[1][1] = __builtin_amdgcn_wmma_f32_16x16x32_bf16(false, atf1, false, bc1, (short)0, aX[1][1], false, false);
        }

        // Online-softmax merge. C/D layout: VGPR j, lane n<16 -> (row j, col n);
        // lane n+16 -> (row j+8, col n). xor masks 1,2,4,8 stay within each half.
#pragma unroll
        for (int ni = 0; ni < 2; ++ni) {
            const float bb = bias[c0 + ni * 16 + l15];
            const float cs = csq [c0 + ni * 16 + l15];
#pragma unroll
            for (int mi = 0; mi < 2; ++mi) {
#pragma unroll
                for (int j = 0; j < 8; ++j) {
                    float logit = aL[mi][ni][j] + bb;
                    float mse   = (tsqr[mi][j] - 2.f * aX[mi][ni][j] + cs) * (1.f / HIDDEN);
                    float mx = logit;
                    mx = fmaxf(mx, __shfl_xor(mx, 1, 32));
                    mx = fmaxf(mx, __shfl_xor(mx, 2, 32));
                    mx = fmaxf(mx, __shfl_xor(mx, 4, 32));
                    mx = fmaxf(mx, __shfl_xor(mx, 8, 32));
                    float mnew = fmaxf(m_run[mi][j], mx);
                    float e  = __expf(logit - mnew);
                    float es = e, en = e * mse;
                    es += __shfl_xor(es, 1, 32); en += __shfl_xor(en, 1, 32);
                    es += __shfl_xor(es, 2, 32); en += __shfl_xor(en, 2, 32);
                    es += __shfl_xor(es, 4, 32); en += __shfl_xor(en, 4, 32);
                    es += __shfl_xor(es, 8, 32); en += __shfl_xor(en, 8, 32);
                    float sc = __expf(m_run[mi][j] - mnew);
                    s_run[mi][j] = s_run[mi][j] * sc + es;
                    n_run[mi][j] = n_run[mi][j] * sc + en;
                    m_run[mi][j] = mnew;
                }
            }
        }
    }

    // Publish per-wave (m, s, num) for its 32 rows.
    if (l15 == 0) {
#pragma unroll
        for (int mi = 0; mi < 2; ++mi)
#pragma unroll
            for (int j = 0; j < 8; ++j) {
                const int r = mi * 16 + half * 8 + j;
                red[w][r][0] = m_run[mi][j];
                red[w][r][1] = s_run[mi][j];
                red[w][r][2] = n_run[mi][j];
            }
    }
    __syncthreads();

    // Merge 8 wave partials per token, apply mask, accumulate scalar loss.
    if (threadIdx.x < 32) {
        const int m = threadIdx.x;
        float M = -3.0e38f;
        for (int ww = 0; ww < 8; ++ww) M = fmaxf(M, red[ww][m][0]);
        float S = 0.f, N = 0.f;
        for (int ww = 0; ww < 8; ++ww) {
            float k = __expf(red[ww][m][0] - M);
            S += red[ww][m][1] * k;
            N += red[ww][m][2] * k;
        }
        const float lossm = N / S;                 // sum_c p_c * mse_c
        const int tt = ttype[m0 + m];
        fin[m] = (tt == 1) ? lossm : 0.f;
    }
    __syncthreads();
    if (threadIdx.x == 0) {
        float s = 0.f;
        for (int i = 0; i < 32; ++i) s += fin[i];
        atomicAdd(out, 0.1f * s);
    }
}

extern "C" void kernel_launch(void* const* d_in, const int* in_sizes, int n_in,
                              void* d_out, int out_size, void* d_ws, size_t ws_size,
                              hipStream_t stream) {
    const float* hidden = (const float*)d_in[0];   // [4,2048,1024]
    const int*   ttype  = (const int*)  d_in[1];   // [4,2048]
    const float* target = (const float*)d_in[2];   // [4,2048,1024]
    const float* cb     = (const float*)d_in[3];   // [8192,1024]
    const float* W      = (const float*)d_in[4];   // [1024,8192]
    const float* b      = (const float*)d_in[5];   // [8192]
    float* out = (float*)d_out;

    // Workspace carve-up (~32.1 MB)
    __bf16* Wt  = (__bf16*)d_ws;                          // [C,H] bf16 (W transposed)
    __bf16* cbb = Wt  + (size_t)CODEBOOK * HIDDEN;        // [C,H] bf16
    float*  csq = (float*)(cbb + (size_t)CODEBOOK * HIDDEN); // [C]
    float*  tsq = csq + CODEBOOK;                         // [NTOK]

    k_zero<<<1, 1, 0, stream>>>(out);
    k_row_convert<<<CODEBOOK, 256, 0, stream>>>(cb, cbb, csq);
    k_row_sq<<<NTOK, 256, 0, stream>>>(target, tsq);
    dim3 tg(CODEBOOK / 32, HIDDEN / 32);
    k_transpose_w<<<tg, 256, 0, stream>>>(W, Wt);
    k_main<<<NTOK / 32, 256, 0, stream>>>(hidden, ttype, target, cbb, Wt, b, csq, tsq, out);
}
